// GRUModel_49581102465312
// MI455X (gfx1250) — compile-verified, matrix operand order checked by reading, not run on hardware
//
#include <hip/hip_runtime.h>
#include <hip/hip_bf16.h>

typedef __attribute__((ext_vector_type(16))) _Float16 v16h;
typedef __attribute__((ext_vector_type(8)))  float    v8f;

#define B_DIM  256
#define T_DIM  288
#define NROW   (B_DIM * T_DIM)   // 73728
#define NHEAD  4
#define HD     64
#define EPSV   1e-5f

// ---------------------------------------------------------------------------
// WMMA helpers (CDNA5 16x16x32 f16 -> f32)
// ---------------------------------------------------------------------------
__device__ __forceinline__ v8f wmma16(v16h a, v16h b, v8f c) {
  // (neg_a, A, neg_b, B, c_mod, C, reuse_a, reuse_b)
  return __builtin_amdgcn_wmma_f32_16x16x32_f16(false, a, false, b, (short)0, c,
                                                false, false);
}

// Fragment load for any operand whose k-dimension is contiguous in memory
// (row-major A [16 x 32], or B given as transposed [16 rows x 32 k]).
// ISA 7.12.2 16-bit layout: lane half h owns k = h*8..h*8+7 (VGPR0-3) and
// k = 16+h*8..16+h*8+7 (VGPR4-7) -> two contiguous 16-byte runs per lane.
__device__ __forceinline__ v16h load_frag_k8(const _Float16* __restrict__ p,
                                             int ld, int lane) {
  int m = lane & 15, h = lane >> 4;
  const _Float16* row = p + (size_t)m * ld + h * 8;
  union { uint4 u[2]; v16h v; } cv;
  cv.u[0] = *(const uint4*)(row);        // k = h*8 .. h*8+7
  cv.u[1] = *(const uint4*)(row + 16);   // k = 16+h*8 .. 16+h*8+7
  return cv.v;
}

__device__ __forceinline__ float sigmoidf_(float x) {
  return 1.0f / (1.0f + __expf(-x));
}

// ---------------------------------------------------------------------------
// Weight preprocessing
// ---------------------------------------------------------------------------
__global__ void k_cvt_t(const float* __restrict__ s, _Float16* __restrict__ d,
                        int K, int N) {  // s[K,N] f32 -> d[N,K] f16 (transposed)
  int i = blockIdx.x * blockDim.x + threadIdx.x;
  if (i < K * N) {
    int k = i / N, n = i - k * N;
    d[(size_t)n * K + k] = (_Float16)s[i];
  }
}

// Activation transpose for attention V: [B*T,256] f16 -> [B,256,T] f16
__global__ void k_transpose_v(const _Float16* __restrict__ src,
                              _Float16* __restrict__ dst) {
  size_t i = (size_t)blockIdx.x * blockDim.x + threadIdx.x;
  if (i >= (size_t)NROW * 256) return;
  int col = (int)(i & 255);
  size_t row = i >> 8;
  int b = (int)(row / T_DIM), t = (int)(row - (size_t)b * T_DIM);
  dst[((size_t)b * 256 + col) * T_DIM + t] = src[i];
}

// ---------------------------------------------------------------------------
// in_dense: [NROW,8] @ [8,256] + b   (K too small for WMMA; VALU)
// ---------------------------------------------------------------------------
__global__ void k_in_dense(const float* __restrict__ cgm,
                           const float* __restrict__ W,
                           const float* __restrict__ b,
                           float* __restrict__ y) {
  int row = blockIdx.x, j = threadIdx.x;
  const float* c = cgm + (size_t)row * 8;
  float acc = b[j];
#pragma unroll
  for (int f = 0; f < 8; ++f) acc = fmaf(c[f], W[f * 256 + j], acc);
  y[(size_t)row * 256 + j] = acc;
}

// ---------------------------------------------------------------------------
// LayerNorm over D=256, optional pre-add / post-add, f32 + f16 outputs
// ---------------------------------------------------------------------------
__global__ void k_ln256(const float* __restrict__ src,
                        const float* __restrict__ preadd,
                        const float* __restrict__ postadd,
                        const float* __restrict__ gamma,
                        const float* __restrict__ beta,
                        float* __restrict__ dstf,
                        _Float16* __restrict__ dsth) {
  __shared__ float red[256];
  int row = blockIdx.x, j = threadIdx.x;
  size_t idx = (size_t)row * 256 + j;
  float x = src[idx];
  if (preadd) x += preadd[idx];
  red[j] = x;
  __syncthreads();
  for (int s = 128; s > 0; s >>= 1) {
    if (j < s) red[j] += red[j + s];
    __syncthreads();
  }
  float mean = red[0] * (1.0f / 256.0f);
  __syncthreads();
  float d = x - mean;
  red[j] = d * d;
  __syncthreads();
  for (int s = 128; s > 0; s >>= 1) {
    if (j < s) red[j] += red[j + s];
    __syncthreads();
  }
  float var = red[0] * (1.0f / 256.0f);
  float o = d * rsqrtf(var + EPSV) * gamma[j] + beta[j];
  if (postadd) o += postadd[idx];
  dstf[idx] = o;
  if (dsth) dsth[idx] = (_Float16)o;
}

// ---------------------------------------------------------------------------
// WMMA GEMM: C[M,N] = A[M,K](f16, row-major) @ Wt[N,K](f16, pre-transposed)
//            + bias.  No LDS: all fragments are 2x b128 global loads, weights
//            stay hot in the 192MB L2.  block = 256 threads (8 waves, 2x4),
//            tile 32(M) x 128(N), each wave owns a 16x32 slab.
// ---------------------------------------------------------------------------
template <bool OUTF16>
__global__ void k_gemm(const _Float16* __restrict__ A,
                       const _Float16* __restrict__ Wt,
                       const float* __restrict__ bias,
                       void* __restrict__ Cout,
                       int M, int N, int K) {
  int tid = threadIdx.x, lane = tid & 31, wave = tid >> 5;
  int wm = wave >> 2, wn = wave & 3;
  int bm = blockIdx.x * 32, bn = blockIdx.y * 128;
  const _Float16* Arow = A + (size_t)(bm + wm * 16) * K;
  const _Float16* Bt0 = Wt + (size_t)(bn + wn * 32) * K;
  const _Float16* Bt1 = Wt + (size_t)(bn + wn * 32 + 16) * K;
  v8f a0 = {}, a1 = {};
  for (int k0 = 0; k0 < K; k0 += 32) {
    v16h af = load_frag_k8(Arow + k0, K, lane);
    v16h b0 = load_frag_k8(Bt0 + k0, K, lane);
    v16h b1 = load_frag_k8(Bt1 + k0, K, lane);
    a0 = wmma16(af, b0, a0);
    a1 = wmma16(af, b1, a1);
  }
  int h = lane >> 4, nl = lane & 15;
#pragma unroll
  for (int rr = 0; rr < 8; ++rr) {
    int row = bm + wm * 16 + rr + 8 * h;
    int c0 = bn + wn * 32 + nl;
    int c1 = c0 + 16;
    float v0 = a0[rr] + bias[c0];
    float v1 = a1[rr] + bias[c1];
    size_t i0 = (size_t)row * N + c0, i1 = (size_t)row * N + c1;
    if (OUTF16) {
      _Float16* C = (_Float16*)Cout;
      C[i0] = (_Float16)v0;
      C[i1] = (_Float16)v1;
    } else {
      float* C = (float*)Cout;
      C[i0] = v0;
      C[i1] = v1;
    }
  }
}

// ---------------------------------------------------------------------------
// GRU scan. 16 workgroups (16 batch rows each), 256 threads = 8 waves.
// h tile [16,256] lives in LDS (f32 + f16 mirror). Per step, wave w computes
// columns n0 = w*32 of h@Whr/Whz/Whn with 48 WMMAs (weights pre-transposed
// [N,K] so B-fragments are 2x b128, L2-resident), then gate math.
// An opaque zero offset is re-materialized by an empty asm each timestep so
// LLVM cannot hoist the (loop-invariant) fragment loads out of the t-loop
// and spill 384 VGPRs of fragments to scratch; the offset is an integer (not
// an inttoptr) so address-space inference still yields global_load_b128.
// ---------------------------------------------------------------------------
__global__ void k_gru(const _Float16* __restrict__ gi,    // [NROW,768] f16
                      const _Float16* __restrict__ WhrT,  // [256,256] (N,K)
                      const _Float16* __restrict__ WhzT,
                      const _Float16* __restrict__ WhnT,
                      const float* __restrict__ bhn,
                      float* __restrict__ y) {            // [NROW,256] f32
  __shared__ float    hs[16 * 256];
  __shared__ _Float16 hh[16 * 256];
  int tid = threadIdx.x, lane = tid & 31, wave = tid >> 5;
  int b0 = blockIdx.x * 16;
  for (int i = tid; i < 16 * 256; i += 256) { hs[i] = 0.0f; hh[i] = (_Float16)0.0f; }
  __syncthreads();
  int n0 = wave * 32;
  unsigned long zoff = 0;  // always 0, but opaque to the optimizer
  for (int t = 0; t < T_DIM; ++t) {
    // opaque offset redefinition: keeps weight fragment loads inside the loop
    asm volatile("" : "+s"(zoff));
    const _Float16* Wmat[3] = {WhrT + zoff, WhzT + zoff, WhnT + zoff};
    v8f acc[3][2];
#pragma unroll
    for (int i = 0; i < 3; ++i) { v8f z = {}; acc[i][0] = z; acc[i][1] = z; }
#pragma unroll
    for (int k = 0; k < 8; ++k) {
      v16h af = load_frag_k8(hh + k * 32, 256, lane);
#pragma unroll
      for (int wmi = 0; wmi < 3; ++wmi) {
        const _Float16* wp0 = Wmat[wmi] + (size_t)n0 * 256 + k * 32;
        v16h bb0 = load_frag_k8(wp0, 256, lane);
        v16h bb1 = load_frag_k8(wp0 + (size_t)16 * 256, 256, lane);
        acc[wmi][0] = wmma16(af, bb0, acc[wmi][0]);
        acc[wmi][1] = wmma16(af, bb1, acc[wmi][1]);
      }
    }
    __syncthreads();
    int h = lane >> 4, nl = lane & 15;
#pragma unroll
    for (int tile = 0; tile < 2; ++tile) {
      int nabs = n0 + tile * 16 + nl;
      float bh = bhn[nabs];
#pragma unroll
      for (int rr = 0; rr < 8; ++rr) {
        int m = rr + 8 * h;
        int brow = b0 + m;
        size_t g = ((size_t)brow * T_DIM + t) * 768;
        float ir  = (float)gi[g + nabs];
        float iz  = (float)gi[g + 256 + nabs];
        float inn = (float)gi[g + 512 + nabs];
        float rg = sigmoidf_(ir + acc[0][tile][rr]);
        float zg = sigmoidf_(iz + acc[1][tile][rr]);
        float ng = tanhf(inn + rg * (acc[2][tile][rr] + bh));
        float hold = hs[m * 256 + nabs];
        float hnew = (1.0f - zg) * ng + zg * hold;
        hs[m * 256 + nabs] = hnew;
        hh[m * 256 + nabs] = (_Float16)hnew;
        y[((size_t)brow * T_DIM + t) * 256 + nabs] = hnew;
      }
    }
    __syncthreads();
  }
}

// ---------------------------------------------------------------------------
// Flash attention: one wave per (batch, head, 16-row q tile).
// Online softmax over 9 kv-chunks of 32.  Q and K fragments are k-contiguous
// in the projection buffers; V is consumed from a [B,256,T] transposed copy
// so P@V B-fragments are also 2x b128.
// ---------------------------------------------------------------------------
__global__ void k_attn(const _Float16* __restrict__ q,   // [B*T,256]
                       const _Float16* __restrict__ k,   // [B*T,256]
                       const _Float16* __restrict__ vT,  // [B,256,T]
                       _Float16* __restrict__ o) {       // [B*T,256]
  int b = blockIdx.x, hd = blockIdx.y, qt = blockIdx.z;
  int lane = threadIdx.x & 31;
  __shared__ _Float16 sp[16 * 32];
  size_t base = (size_t)b * T_DIM * 256 + hd * HD;
  size_t qrow = base + (size_t)(qt * 16) * 256;
  v16h aq0 = load_frag_k8(q + qrow, 256, lane);
  v16h aq1 = load_frag_k8(q + qrow + 32, 256, lane);
  const _Float16* vtb = vT + ((size_t)b * 256 + hd * HD) * T_DIM;
  v8f o0 = {}, o1 = {}, o2 = {}, o3 = {};
  float mrun[8], lrun[8];
#pragma unroll
  for (int rr = 0; rr < 8; ++rr) { mrun[rr] = -1e30f; lrun[rr] = 0.0f; }
  for (int c = 0; c < T_DIM / 32; ++c) {
    size_t kv = base + (size_t)(c * 32) * 256;
    v8f s0 = {}, s1 = {};
    // scores tile: B[kd][n] = K[kv_row n][kd] -> k-contiguous at k+kv+n*256
    v16h b00 = load_frag_k8(k + kv, 256, lane);
    v16h b01 = load_frag_k8(k + kv + 32, 256, lane);
    v16h b10 = load_frag_k8(k + kv + (size_t)16 * 256, 256, lane);
    v16h b11 = load_frag_k8(k + kv + (size_t)16 * 256 + 32, 256, lane);
    s0 = wmma16(aq0, b00, s0);
    s0 = wmma16(aq1, b01, s0);
    s1 = wmma16(aq0, b10, s1);
    s1 = wmma16(aq1, b11, s1);
    float alpha[8], p0[8], p1[8];
#pragma unroll
    for (int rr = 0; rr < 8; ++rr) {
      float sv0 = s0[rr] * 0.125f, sv1 = s1[rr] * 0.125f;
      float mx = fmaxf(sv0, sv1);
      mx = fmaxf(mx, __shfl_xor(mx, 1, 32));
      mx = fmaxf(mx, __shfl_xor(mx, 2, 32));
      mx = fmaxf(mx, __shfl_xor(mx, 4, 32));
      mx = fmaxf(mx, __shfl_xor(mx, 8, 32));
      float mn = fmaxf(mrun[rr], mx);
      float a = __expf(mrun[rr] - mn);
      float e0 = __expf(sv0 - mn), e1 = __expf(sv1 - mn);
      float ss = e0 + e1;
      ss += __shfl_xor(ss, 1, 32);
      ss += __shfl_xor(ss, 2, 32);
      ss += __shfl_xor(ss, 4, 32);
      ss += __shfl_xor(ss, 8, 32);
      lrun[rr] = lrun[rr] * a + ss;
      mrun[rr] = mn;
      alpha[rr] = a; p0[rr] = e0; p1[rr] = e1;
    }
#pragma unroll
    for (int rr = 0; rr < 8; ++rr) {
      o0[rr] *= alpha[rr]; o1[rr] *= alpha[rr];
      o2[rr] *= alpha[rr]; o3[rr] *= alpha[rr];
    }
    int h = lane >> 4, nl = lane & 15;
#pragma unroll
    for (int rr = 0; rr < 8; ++rr) {
      sp[(rr + 8 * h) * 32 + nl]      = (_Float16)p0[rr];
      sp[(rr + 8 * h) * 32 + 16 + nl] = (_Float16)p1[rr];
    }
    __syncthreads();
    v16h ap  = load_frag_k8(sp, 32, lane);
    // P@V: B[kk][n] = V[kv_t kk][col n] = vT[col n][kk] -> k-contiguous
    v16h bv0 = load_frag_k8(vtb + c * 32, T_DIM, lane);
    v16h bv1 = load_frag_k8(vtb + (size_t)16 * T_DIM + c * 32, T_DIM, lane);
    v16h bv2 = load_frag_k8(vtb + (size_t)32 * T_DIM + c * 32, T_DIM, lane);
    v16h bv3 = load_frag_k8(vtb + (size_t)48 * T_DIM + c * 32, T_DIM, lane);
    o0 = wmma16(ap, bv0, o0);
    o1 = wmma16(ap, bv1, o1);
    o2 = wmma16(ap, bv2, o2);
    o3 = wmma16(ap, bv3, o3);
    __syncthreads();
  }
  int h = lane >> 4, nl = lane & 15;
#pragma unroll
  for (int rr = 0; rr < 8; ++rr) {
    float inv = 1.0f / lrun[rr];
    size_t row = qrow + (size_t)(rr + 8 * h) * 256;
    o[row + nl]      = (_Float16)(o0[rr] * inv);
    o[row + 16 + nl] = (_Float16)(o1[rr] * inv);
    o[row + 32 + nl] = (_Float16)(o2[rr] * inv);
    o[row + 48 + nl] = (_Float16)(o3[rr] * inv);
  }
}

// ---------------------------------------------------------------------------
// Head: temporal mean pool + concat, dense+relu+LN, final out
// ---------------------------------------------------------------------------
__global__ void k_pool_concat(const float* __restrict__ x,
                              const float* __restrict__ other,
                              float* __restrict__ comb) {
  int b = blockIdx.x, j = threadIdx.x;  // 272 threads
  if (j < 256) {
    float s = 0.0f;
    for (int t = 0; t < T_DIM; ++t) s += x[((size_t)b * T_DIM + t) * 256 + j];
    comb[b * 272 + j] = s * (1.0f / T_DIM);
  } else {
    comb[b * 272 + j] = other[b * 16 + (j - 256)];
  }
}

__global__ void k_dense_relu_ln(const float* __restrict__ in,
                                const float* __restrict__ W,
                                const float* __restrict__ bias,
                                const float* __restrict__ gamma,
                                const float* __restrict__ beta,
                                float* __restrict__ out, int din, int dout) {
  __shared__ float red[128];
  int row = blockIdx.x, j = threadIdx.x;  // blockDim == dout (128 or 64)
  float acc = bias[j];
  for (int i = 0; i < din; ++i) acc = fmaf(in[row * din + i], W[i * dout + j], acc);
  acc = fmaxf(acc, 0.0f);
  red[j] = acc;
  __syncthreads();
  for (int s = dout >> 1; s > 0; s >>= 1) {
    if (j < s) red[j] += red[j + s];
    __syncthreads();
  }
  float mean = red[0] / (float)dout;
  __syncthreads();
  float d = acc - mean;
  red[j] = d * d;
  __syncthreads();
  for (int s = dout >> 1; s > 0; s >>= 1) {
    if (j < s) red[j] += red[j + s];
    __syncthreads();
  }
  float var = red[0] / (float)dout;
  out[row * dout + j] = d * rsqrtf(var + EPSV) * gamma[j] + beta[j];
}

__global__ void k_head_out(const float* __restrict__ h,
                           const float* __restrict__ W,
                           const float* __restrict__ b,
                           float* __restrict__ out) {
  int bi = blockIdx.x * blockDim.x + threadIdx.x;  // 256 batch rows
  float s = b[0];
#pragma unroll
  for (int i = 0; i < 64; ++i) s = fmaf(h[bi * 64 + i], W[i], s);
  out[bi] = s;
}

// ---------------------------------------------------------------------------
// Host launch
// ---------------------------------------------------------------------------
extern "C" void kernel_launch(void* const* d_in, const int* in_sizes, int n_in,
                              void* d_out, int out_size, void* d_ws, size_t ws_size,
                              hipStream_t stream) {
  (void)in_sizes; (void)n_in; (void)out_size; (void)ws_size;
  auto in = [&](int i) -> const float* { return (const float*)d_in[i]; };
  const float* cgm = in(0);
  const float* oth = in(1);
  // params flattened by sorted dict keys:
  // blocks[b]: attn.k.{W,b}, attn.o.{W,b}, attn.q.{W,b}, attn.v.{W,b},
  //            gru.{Whn,Whr,Whz,Wi,bhn,bi}, ln1.{bias,scale}, ln2.{bias,scale}
  // then d128.{W,b}, d64.{W,b}, in_dense.{W,b}, in_ln.{bias,scale},
  //      ln128.{bias,scale}, ln64.{bias,scale}, out.{W,b}

  char* ws = (char*)d_ws;
  size_t off = 0;
  auto alloc = [&](size_t bytes) -> void* {
    void* p = ws + off;
    off += (bytes + 255) & ~(size_t)255;
    return p;
  };
  float*    xf32 = (float*)alloc((size_t)NROW * 256 * 4);
  _Float16* xf16 = (_Float16*)alloc((size_t)NROW * 256 * 2);
  float*    yf32 = (float*)alloc((size_t)NROW * 256 * 4);
  _Float16* gi16 = (_Float16*)alloc((size_t)NROW * 768 * 2);
  _Float16* q16  = (_Float16*)alloc((size_t)NROW * 256 * 2);
  _Float16* kk16 = (_Float16*)alloc((size_t)NROW * 256 * 2);
  _Float16* vv16 = (_Float16*)alloc((size_t)NROW * 256 * 2);
  _Float16* vT16 = (_Float16*)alloc((size_t)NROW * 256 * 2);
  _Float16* at16 = (_Float16*)alloc((size_t)NROW * 256 * 2);
  _Float16* w16[2][8];  // WiT, WhrT, WhzT, WhnT, WqT, WkT, WvT, WoT  (all [N,K])
  for (int blk = 0; blk < 2; ++blk) {
    w16[blk][0] = (_Float16*)alloc((size_t)256 * 768 * 2);
    for (int i = 1; i < 8; ++i) w16[blk][i] = (_Float16*)alloc((size_t)256 * 256 * 2);
  }
  float* comb  = (float*)alloc((size_t)256 * 272 * 4);
  float* h128b = (float*)alloc((size_t)256 * 128 * 4);
  float* h64b  = (float*)alloc((size_t)256 * 64 * 4);

  auto cvtT = [&](const float* s, _Float16* d, int K, int N) {
    k_cvt_t<<<(K * N + 255) / 256, 256, 0, stream>>>(s, d, K, N);
  };
  for (int blk = 0; blk < 2; ++blk) {
    int base = 2 + blk * 18;
    cvtT(in(base + 11), w16[blk][0], 256, 768);  // Wi
    cvtT(in(base + 9),  w16[blk][1], 256, 256);  // Whr
    cvtT(in(base + 10), w16[blk][2], 256, 256);  // Whz
    cvtT(in(base + 8),  w16[blk][3], 256, 256);  // Whn
    cvtT(in(base + 4),  w16[blk][4], 256, 256);  // Wq
    cvtT(in(base + 0),  w16[blk][5], 256, 256);  // Wk
    cvtT(in(base + 6),  w16[blk][6], 256, 256);  // Wv
    cvtT(in(base + 2),  w16[blk][7], 256, 256);  // Wo
  }

  // in_dense + in_ln
  k_in_dense<<<NROW, 256, 0, stream>>>(cgm, in(42), in(43), yf32);
  k_ln256<<<NROW, 256, 0, stream>>>(yf32, nullptr, nullptr, in(45), in(44), xf32, xf16);

  for (int blk = 0; blk < 2; ++blk) {
    int base = 2 + blk * 18;
    // gi = x @ Wi + bi   (f16 out)
    dim3 g1(NROW / 32, 768 / 128);
    k_gemm<true><<<g1, 256, 0, stream>>>(xf16, w16[blk][0], in(base + 13), gi16,
                                         NROW, 768, 256);
    // GRU scan
    k_gru<<<16, 256, 0, stream>>>(gi16, w16[blk][1], w16[blk][2], w16[blk][3],
                                  in(base + 12), yf32);
    // x = LN(gru) + skip1   (post-add)
    k_ln256<<<NROW, 256, 0, stream>>>(yf32, nullptr, xf32, in(base + 15), in(base + 14),
                                      xf32, xf16);
    // Q,K,V projections (f16 out)
    dim3 g2(NROW / 32, 256 / 128);
    k_gemm<true><<<g2, 256, 0, stream>>>(xf16, w16[blk][4], in(base + 5), q16,
                                         NROW, 256, 256);
    k_gemm<true><<<g2, 256, 0, stream>>>(xf16, w16[blk][5], in(base + 1), kk16,
                                         NROW, 256, 256);
    k_gemm<true><<<g2, 256, 0, stream>>>(xf16, w16[blk][6], in(base + 7), vv16,
                                         NROW, 256, 256);
    // transpose V for k-contiguous P@V fragments
    k_transpose_v<<<(int)(((size_t)NROW * 256 + 255) / 256), 256, 0, stream>>>(vv16, vT16);
    // attention
    dim3 g3(B_DIM, NHEAD, T_DIM / 16);
    k_attn<<<g3, 32, 0, stream>>>(q16, kk16, vT16, at16);
    // O projection (f32 out)
    k_gemm<false><<<g2, 256, 0, stream>>>(at16, w16[blk][7], in(base + 3), yf32,
                                          NROW, 256, 256);
    // x = LN(attn_out + skip2)   (pre-add)
    k_ln256<<<NROW, 256, 0, stream>>>(yf32, xf32, nullptr, in(base + 17), in(base + 16),
                                      xf32, xf16);
  }

  // head
  k_pool_concat<<<B_DIM, 272, 0, stream>>>(xf32, oth, comb);
  k_dense_relu_ln<<<B_DIM, 128, 0, stream>>>(comb, in(38), in(39), in(47), in(46),
                                             h128b, 272, 128);
  k_dense_relu_ln<<<B_DIM, 64, 0, stream>>>(h128b, in(40), in(41), in(49), in(48),
                                            h64b, 128, 64);
  k_head_out<<<1, 256, 0, stream>>>(h64b, in(50), in(51), (float*)d_out);
}
